// SinusoidPositionEncoding_67954972557888
// MI455X (gfx1250) — matrix operand order repaired and checked by reading, not compiled, not run on hardware
//
#include <hip/hip_runtime.h>

// Problem dims (fixed by the reference): B=8, S=2048, MAX_LEN=4096, EMBED_DIM=128
#define MAXLEN   4096
#define DDIM     128
#define MTOT     16384  // B*S
#define PF_NEAR  4      // shallow prefetch depth (k-iterations)
#define PF_FAR   16     // deep prefetch depth (k-iterations): GL2

typedef __attribute__((ext_vector_type(16))) __bf16 v16bf;
typedef __attribute__((ext_vector_type(8)))  float  v8f;

// Round-to-nearest-even f32 -> bf16 (raw u16). Used for the PE split, where
// precision matters.
__device__ __forceinline__ unsigned short bf16_rne(float f) {
  unsigned int u = __float_as_uint(f);
  u += 0x7fffu + ((u >> 16) & 1u);
  return (unsigned short)(u >> 16);
}

// Truncating pack of two f32 into two bf16 in one dword. Exact for values that
// are bf16-representable — the one-hot x is {0.0, 1.0}, so this is bit-exact.
__device__ __forceinline__ unsigned int pack2_bf16_trunc(float f0, float f1) {
  return (__float_as_uint(f1) & 0xffff0000u) | (__float_as_uint(f0) >> 16);
}

// Prep: split pos_enc into bf16 hi + bf16 residual, stored TRANSPOSED (d-major)
// so WMMA B-fragments become contiguous 32B loads along K.
__global__ __launch_bounds__(256) void pe_split_T(const float* __restrict__ pe,
                                                  unsigned short* __restrict__ hiT,
                                                  unsigned short* __restrict__ loT) {
  int i = blockIdx.x * blockDim.x + threadIdx.x;
  if (i >= MAXLEN * DDIM) return;
  int m = i / DDIM;
  int d = i - m * DDIM;
  float v = pe[i];
  unsigned short h = bf16_rne(v);
  float hf = __uint_as_float(((unsigned int)h) << 16);
  unsigned short l = bf16_rne(v - hf);
  hiT[(size_t)d * MAXLEN + m] = h;
  loT[(size_t)d * MAXLEN + m] = l;
}

// GEMM: out(16384x128) = X(16384x4096,f32) * PE(4096x128)
// One wave computes one 16-row x 128-col slab (8 N-tiles of 16x16).
// 4 waves/block, 64 rows/block -> 256 blocks, 1024 waves.
__global__ __launch_bounds__(128) void onehot_pe_wmma(const float* __restrict__ x,
                                                      const unsigned short* __restrict__ hiT,
                                                      const unsigned short* __restrict__ loT,
                                                      float* __restrict__ out) {
  const int lane  = threadIdx.x & 31;
  const int wave  = threadIdx.x >> 5;
  const int mtile = blockIdx.x * 4 + wave;   // 16-row tile index (0..1023)
  const int nlane = lane & 15;               // row within M-tile / col within N-tile
  const int khalf = lane >> 4;               // 0 or 1 (lane-half selects K sub-block)

  v8f acc[8];
#pragma unroll
  for (int nt = 0; nt < 8; ++nt) acc[nt] = (v8f)(0.0f);

  // A-fragment base (16-bit A 16x32, this code's contiguous K order, same as B):
  // lane<16 holds K {0..15}; lane>=16 holds K {16..31}.
  const float* aptr = x + ((size_t)mtile * 16 + nlane) * MAXLEN + khalf * 16;

  // Prologue: raw A for kb=0 (software-pipelined one iteration ahead).
  float4 c0 = *(const float4*)(aptr);
  float4 c1 = *(const float4*)(aptr + 4);
  float4 c2 = *(const float4*)(aptr + 8);
  float4 c3 = *(const float4*)(aptr + 12);

  for (int kb = 0; kb < MAXLEN; kb += 32) {
    // ---- Issue ALL B-fragment loads (16-bit B 32x16 in this code's contiguous K order: lane = column,
    // lane-half = K half; 16 contiguous bf16 along K in transposed PE arrays).
    v16bf bh[8], bl[8];
    const size_t boff = (size_t)nlane * MAXLEN + (size_t)kb + (size_t)khalf * 16;
#pragma unroll
    for (int nt = 0; nt < 8; ++nt)
      bh[nt] = *(const v16bf*)(hiT + boff + (size_t)nt * 16 * MAXLEN);
#pragma unroll
    for (int nt = 0; nt < 8; ++nt)
      bl[nt] = *(const v16bf*)(loT + boff + (size_t)nt * 16 * MAXLEN);

    // ---- Issue NEXT iteration's raw A loads (pointer clamped at the tail so
    // demand loads never run past x; prefetches are speculative by spec).
    const int kn = (kb + 32 < MAXLEN) ? (kb + 32) : kb;
    const float* apn = aptr + kn;
    float4 d0 = *(const float4*)(apn);
    float4 d1 = *(const float4*)(apn + 4);
    float4 d2 = *(const float4*)(apn + 8);
    float4 d3 = *(const float4*)(apn + 12);
    // Each lane's A slice advances exactly one 128B line per k-iteration.
    __builtin_prefetch(apn + 32 * (PF_NEAR - 1), 0, 0);
    __builtin_prefetch(apn + 32 * (PF_FAR - 1), 0, 3);

    // ---- Convert CURRENT A (loaded a full iteration ago -> wait already paid).
    union { v16bf v; unsigned int u[8]; } af;
    af.u[0] = pack2_bf16_trunc(c0.x, c0.y);
    af.u[1] = pack2_bf16_trunc(c0.z, c0.w);
    af.u[2] = pack2_bf16_trunc(c1.x, c1.y);
    af.u[3] = pack2_bf16_trunc(c1.z, c1.w);
    af.u[4] = pack2_bf16_trunc(c2.x, c2.y);
    af.u[5] = pack2_bf16_trunc(c2.z, c2.w);
    af.u[6] = pack2_bf16_trunc(c3.x, c3.y);
    af.u[7] = pack2_bf16_trunc(c3.z, c3.w);

    // ---- Hard scheduling fence: loads/conversion may NOT sink past here, so
    // all 16 B fragments stay simultaneously live -> batched clause + staged
    // waits, then back-to-back WMMAs instead of load->wait(0)->wmma chains.
    __builtin_amdgcn_sched_barrier(0);

#pragma unroll
    for (int nt = 0; nt < 8; ++nt) {
      acc[nt] = __builtin_amdgcn_wmma_f32_16x16x32_bf16(
          false, af.v, false, bh[nt], (short)0, acc[nt], false, false);
    }
#pragma unroll
    for (int nt = 0; nt < 8; ++nt) {
      acc[nt] = __builtin_amdgcn_wmma_f32_16x16x32_bf16(
          false, af.v, false, bl[nt], (short)0, acc[nt], false, false);
    }

    // ---- Rotate the A pipeline.
    c0 = d0; c1 = d1; c2 = d2; c3 = d3;
  }

  // Store: C/D layout — lane 0..15: (M=r, N=lane); lane 16..31: (M=8+r, N=lane-16).
  const size_t rbase = (size_t)mtile * 16 + khalf * 8;
#pragma unroll
  for (int nt = 0; nt < 8; ++nt) {
#pragma unroll
    for (int r = 0; r < 8; ++r) {
      out[(rbase + r) * DDIM + nt * 16 + nlane] = acc[nt][r];
    }
  }
}

extern "C" void kernel_launch(void* const* d_in, const int* in_sizes, int n_in,
                              void* d_out, int out_size, void* d_ws, size_t ws_size,
                              hipStream_t stream) {
  const float* x  = (const float*)d_in[0];   // (8, 2048, 4096) f32 one-hot
  const float* pe = (const float*)d_in[1];   // (4096, 128) f32
  float* out = (float*)d_out;                // (8, 2048, 128) f32

  unsigned short* hiT = (unsigned short*)d_ws;            // 128*4096 bf16 (1 MB)
  unsigned short* loT = hiT + (size_t)MAXLEN * DDIM;      // 128*4096 bf16 (1 MB)

  pe_split_T<<<(MAXLEN * DDIM + 255) / 256, 256, 0, stream>>>(pe, hiT, loT);
  onehot_pe_wmma<<<MTOT / 64, 128, 0, stream>>>(x, hiT, loT, out);
}
